// MLA_12979391168583
// MI455X (gfx1250) — matrix operand order repaired and checked by reading, not compiled, not run on hardware
//
#include <hip/hip_runtime.h>

// Problem constants (from reference): B=2, S=1024, HS=1024, NH=16, L=3, HD=64
#define LVL 3
#define NB 2
#define SEQ 1024
#define HSZ 1024
#define NHD 16
#define HDM 64

typedef __attribute__((ext_vector_type(16))) __bf16 v16bf;
typedef __attribute__((ext_vector_type(8))) __bf16 v8bf;
typedef __attribute__((ext_vector_type(8))) float v8f;
typedef __attribute__((ext_vector_type(8))) float v8fl;
typedef __attribute__((__vector_size__(4 * sizeof(int)))) int v4i;

#if defined(__has_builtin)
#if __has_builtin(__builtin_amdgcn_global_load_async_to_lds_b128) && \
    __has_builtin(__builtin_amdgcn_s_wait_asynccnt)
#define USE_ASYNC_LDS 1
#endif
#endif
#ifndef USE_ASYNC_LDS
#define USE_ASYNC_LDS 0
#endif

#if USE_ASYNC_LDS
__device__ __forceinline__ void async_b128(const void* g, void* l) {
  __builtin_amdgcn_global_load_async_to_lds_b128(
      (__attribute__((address_space(1))) v4i*)(g),
      (__attribute__((address_space(3))) v4i*)(l), 0, 0);
}
#define ASYNC_WAIT_ALL() __builtin_amdgcn_s_wait_asynccnt(0)
#endif

union FragBf {
  v16bf v;
  uint4 q[2];
};

__device__ __forceinline__ v8f vzero8() {
  v8f z;
#pragma unroll
  for (int i = 0; i < 8; ++i) z[i] = 0.f;
  return z;
}

// hardware f32 -> bf16 (v_cvt_pk_bf16_f32 path via scalar __bf16 cast)
__device__ __forceinline__ unsigned short f2bf(float x) {
  union {
    __bf16 h;
    unsigned short u;
  } c;
  c.h = (__bf16)x;
  return c.u;
}

union Pack8U {
  v8bf b;
  uint4 q;
};

__device__ __forceinline__ uint4 pack8(float4 x, float4 y) {
  v8fl f;
  f[0] = x.x; f[1] = x.y; f[2] = x.z; f[3] = x.w;
  f[4] = y.x; f[5] = y.y; f[6] = y.z; f[7] = y.w;
  Pack8U u;
  u.b = __builtin_convertvector(f, v8bf);
  return u.q;
}

__device__ __forceinline__ v8f wmma_bf16(const FragBf& a, const FragBf& b, v8f c) {
  return __builtin_amdgcn_wmma_f32_16x16x32_bf16(false, a.v, false, b.v, (short)0,
                                                 c, false, false);
}

__device__ __forceinline__ float red_max16(float x) {
#pragma unroll
  for (int off = 8; off > 0; off >>= 1) x = fmaxf(x, __shfl_xor(x, off, 32));
  return x;
}
__device__ __forceinline__ float red_sum16(float x) {
#pragma unroll
  for (int off = 8; off > 0; off >>= 1) x += __shfl_xor(x, off, 32);
  return x;
}

// ---------------------------------------------------------------------------
// One-time f32 -> bf16 conversion (streaming, packed cvt)
// ---------------------------------------------------------------------------
__global__ __launch_bounds__(256) void mla_cvt_bf16(const float* __restrict__ in,
                                                    unsigned short* __restrict__ out,
                                                    int n8) {
  const int i = blockIdx.x * 256 + threadIdx.x;
  if (i < n8) {
    const float4* p = (const float4*)(in + (size_t)i * 8);
    *(uint4*)(out + (size_t)i * 8) = pack8(p[0], p[1]);
  }
}

// ---------------------------------------------------------------------------
// bf16 WMMA GEMM:  out[m,n] = sum_k A[m,k] * W[n,k] + bias[n]
// Block tile 128x128, 8 waves (2x4), wave tile 64x32.
// K staged 32/step through double-buffered LDS via async global->LDS b128.
// batched (grid.z = 9): z -> (level l, op); op 0=Q, 1=K (bf16), 2=V (bf16,
// transposed per (l,b) as [d][s]).  fp32_out: final projection, f32 out.
// ---------------------------------------------------------------------------
__global__ __launch_bounds__(256) void mla_gemm_wmma(
    const unsigned short* __restrict__ A, const unsigned short* __restrict__ W0,
    const unsigned short* __restrict__ W1, const unsigned short* __restrict__ W2,
    const float* __restrict__ b0, const float* __restrict__ b1,
    const float* __restrict__ b2, unsigned short* out0, unsigned short* out1,
    unsigned short* out2, float* outf, int M, int N, int K, int batched,
    int fp32_out) {
  // 80B row stride (32 data halves + 8 pad): 16B aligned, bank-spread
  __shared__ __align__(16) unsigned lA[2][128 * 20];
  __shared__ __align__(16) unsigned lB[2][128 * 20];

  const int t = threadIdx.x;
  const int m0 = blockIdx.y * 128;
  const int n0 = blockIdx.x * 128;
  const int z = blockIdx.z;
  const int l = batched ? (z / 3) : 0;
  const int op = batched ? (z - 3 * l) : 0;

  const unsigned short* W =
      (op == 0 ? W0 : (op == 1 ? W1 : W2)) + (size_t)l * N * K;
  const float* bias = (op == 0 ? b0 : (op == 1 ? b1 : b2)) + (size_t)l * N;
  unsigned short* outh = (op == 0 ? out0 : (op == 1 ? out1 : out2));
  const size_t lvl_off = (size_t)l * M * N;
  const int mode = fp32_out ? 2 : (op == 2 ? 1 : 0);

  const int lane = t & 31, wid = t >> 5;
  const int wm = wid >> 2, wn = wid & 3;
  const int ml = lane & 15, hf = lane >> 4;

  v8f acc[4][2];
#pragma unroll
  for (int mi = 0; mi < 4; ++mi)
#pragma unroll
    for (int ni = 0; ni < 2; ++ni) acc[mi][ni] = vzero8();

  // staging map: 256 threads x 2 segments cover 128 rows x 32 halves (16B segs)
  const int rowT = t >> 2;  // 0..63 (+64 for segment 1)
  const int quad = t & 3;

  auto computeTile = [&](int buf) {
    FragBf a[4], bw[2];
#pragma unroll
    for (int mi = 0; mi < 4; ++mi) {
      const unsigned* base = &lA[buf][(wm * 64 + mi * 16 + ml) * 20];
      a[mi].q[0] = *(const uint4*)(base + hf * 4);
      a[mi].q[1] = *(const uint4*)(base + 8 + hf * 4);
    }
#pragma unroll
    for (int ni = 0; ni < 2; ++ni) {
      const unsigned* base = &lB[buf][(wn * 32 + ni * 16 + ml) * 20];
      bw[ni].q[0] = *(const uint4*)(base + hf * 4);
      bw[ni].q[1] = *(const uint4*)(base + 8 + hf * 4);
    }
#pragma unroll
    for (int mi = 0; mi < 4; ++mi)
#pragma unroll
      for (int ni = 0; ni < 2; ++ni)
        acc[mi][ni] = wmma_bf16(a[mi], bw[ni], acc[mi][ni]);
  };

  const int KT = K >> 5;

#if USE_ASYNC_LDS
  auto issueTile = [&](int kt, int buf) {
#pragma unroll
    for (int i = 0; i < 2; ++i) {
      const int row = rowT + 64 * i;
      const size_t koff = (size_t)kt * 32 + quad * 8;
      async_b128(A + (size_t)(m0 + row) * K + koff,
                 &lA[buf][row * 20 + quad * 4]);
      async_b128(W + (size_t)(n0 + row) * K + koff,
                 &lB[buf][row * 20 + quad * 4]);
    }
  };
  issueTile(0, 0);
#pragma unroll 1
  for (int kt = 0; kt < KT; ++kt) {
    ASYNC_WAIT_ALL();
    __syncthreads();
    if (kt + 1 < KT) issueTile(kt + 1, (kt + 1) & 1);
    computeTile(kt & 1);
  }
#else
#pragma unroll 1
  for (int kt = 0; kt < KT; ++kt) {
    uint4 r[4];
#pragma unroll
    for (int i = 0; i < 2; ++i) {
      const int row = rowT + 64 * i;
      const size_t koff = (size_t)kt * 32 + quad * 8;
      r[i] = *(const uint4*)(A + (size_t)(m0 + row) * K + koff);
      r[2 + i] = *(const uint4*)(W + (size_t)(n0 + row) * K + koff);
    }
#pragma unroll
    for (int i = 0; i < 2; ++i) {
      const int row = rowT + 64 * i;
      *(uint4*)&lA[0][row * 20 + quad * 4] = r[i];
      *(uint4*)&lB[0][row * 20 + quad * 4] = r[2 + i];
    }
    __syncthreads();
    computeTile(0);
    __syncthreads();
  }
#endif

  // epilogue: C layout = lane(col) x vgpr(row), rows split by lane half
#pragma unroll
  for (int mi = 0; mi < 4; ++mi) {
#pragma unroll
    for (int ni = 0; ni < 2; ++ni) {
      const int ncol = n0 + wn * 32 + ni * 16 + ml;
      const float bv = bias[ncol];
      const int rbase = m0 + wm * 64 + mi * 16 + hf * 8;
#pragma unroll
      for (int v = 0; v < 8; ++v) {
        const int gm = rbase + v;
        const float val = acc[mi][ni][v] + bv;
        if (mode == 2) {
          outf[(size_t)gm * N + ncol] = val;
        } else if (mode == 1) {
          const size_t addr =
              ((size_t)(l * NB + (gm >> 10)) * N + ncol) * SEQ + (gm & (SEQ - 1));
          outh[addr] = f2bf(val);
        } else {
          outh[lvl_off + (size_t)gm * N + ncol] = f2bf(val);
        }
      }
    }
  }
}

// ---------------------------------------------------------------------------
// Flash-style attention per (level, batch, head): wave = 16 query rows,
// keys streamed 32/step. K (32x64) and V^T (64x32) tiles staged in LDS via
// async copies, double buffered; Q.K^T and P.V both bf16 WMMA, f32 acc.
// ---------------------------------------------------------------------------
__global__ __launch_bounds__(256) void mla_attn_wmma(
    const unsigned short* __restrict__ Qb, const unsigned short* __restrict__ Kb,
    const unsigned short* __restrict__ Vt, const unsigned char* __restrict__ mask,
    unsigned short* __restrict__ ctxout) {
  __shared__ float smask[SEQ];
  __shared__ __align__(16) unsigned short sP[8 * 16 * 40];  // wave-private P
#if USE_ASYNC_LDS
  __shared__ __align__(16) unsigned short sK[2][32 * 72];  // 72-half row stride
  __shared__ __align__(16) unsigned short sV[2][64 * 40];  // 40-half row stride
#endif

  const int t = threadIdx.x, lane = t & 31, wid = t >> 5;
  const int ml = lane & 15, hf = lane >> 4;
  const int z = blockIdx.z;
  const int l = z >> 1, b = z & 1;
  const int h = blockIdx.y;
  const int qb = blockIdx.x * 128 + wid * 16;
  const float NEG_INF = -__builtin_inff();

  const unsigned char* mrow = mask + b * SEQ;
#pragma unroll
  for (int j = 0; j < 4; ++j) {
    const int key = t * 4 + j;
    smask[key] = mrow[key] ? NEG_INF : 0.f;
  }

  const float sc = (float)(1 << l) * 0.125f;  // 2^l * HD^-0.5
  const size_t lb_row = (size_t)(l * (NB * SEQ) + b * SEQ);

  FragBf aq[2];
  {
    const unsigned short* qp = Qb + (lb_row + qb + ml) * HSZ + h * HDM;
#pragma unroll
    for (int ks = 0; ks < 2; ++ks) {
      aq[ks].q[0] = *(const uint4*)(qp + ks * 32 + hf * 8);
      aq[ks].q[1] = *(const uint4*)(qp + ks * 32 + 16 + hf * 8);
    }
  }

  float rmax[8], rsum[8];
  v8f oacc[4];
#pragma unroll
  for (int v = 0; v < 8; ++v) {
    rmax[v] = NEG_INF;
    rsum[v] = 0.f;
  }
#pragma unroll
  for (int j = 0; j < 4; ++j) oacc[j] = vzero8();

  const unsigned short* kbase = Kb + lb_row * HSZ + h * HDM;
  const size_t vt_base = ((size_t)(l * NB + b) * HSZ + h * HDM) * SEQ;
  unsigned short* wp = &sP[wid * 640];

#if USE_ASYNC_LDS
  auto issueKV = [&](int kb, int buf) {
    {  // K tile: 32 rows x 128B; thread -> (row = t>>3, 16B quad = t&7)
      const int row = t >> 3, q8 = t & 7;
      async_b128(kbase + (size_t)(kb + row) * HSZ + q8 * 8,
                 &sK[buf][row * 72 + q8 * 8]);
    }
    {  // V^T tile: 64 rows x 64B; thread -> (row = t>>2, 16B quad = t&3)
      const int row = t >> 2, q4 = t & 3;
      async_b128(Vt + vt_base + (size_t)row * SEQ + kb + q4 * 8,
                 &sV[buf][row * 40 + q4 * 8]);
    }
  };
  issueKV(0, 0);
#endif
  __syncthreads();

#pragma unroll 1
  for (int kb = 0; kb < SEQ; kb += 32) {
#if USE_ASYNC_LDS
    const int buf = (kb >> 5) & 1;
    ASYNC_WAIT_ALL();
    __syncthreads();
    if (kb + 32 < SEQ) issueKV(kb + 32, buf ^ 1);
#endif

    // ---- S = Q.K^T ----
    FragBf kf[2][2];
#pragma unroll
    for (int tt = 0; tt < 2; ++tt) {
#if USE_ASYNC_LDS
      const unsigned short* kp = &sK[buf][(tt * 16 + ml) * 72];
#else
      const unsigned short* kp = kbase + (size_t)(kb + tt * 16 + ml) * HSZ;
#endif
#pragma unroll
      for (int ks = 0; ks < 2; ++ks) {
        kf[tt][ks].q[0] = *(const uint4*)(kp + ks * 32 + hf * 8);
        kf[tt][ks].q[1] = *(const uint4*)(kp + ks * 32 + 16 + hf * 8);
      }
    }
    v8f st[2];
#pragma unroll
    for (int tt = 0; tt < 2; ++tt) {
      v8f c = vzero8();
      c = wmma_bf16(aq[0], kf[tt][0], c);
      c = wmma_bf16(aq[1], kf[tt][1], c);
      st[tt] = c;
    }

    // ---- online softmax ----
    float p0[8], p1[8], mnew[8];
#pragma unroll
    for (int v = 0; v < 8; ++v) {
      const float s0 = st[0][v] * sc + smask[kb + ml];
      const float s1 = st[1][v] * sc + smask[kb + 16 + ml];
      p0[v] = s0;
      p1[v] = s1;
      mnew[v] = fmaxf(rmax[v], red_max16(fmaxf(s0, s1)));
    }
#pragma unroll
    for (int v = 0; v < 8; ++v) {
      const float mn = mnew[v];
      const float corr = (rmax[v] == NEG_INF) ? 0.f : __expf(rmax[v] - mn);
      const float e0 = (mn == NEG_INF) ? 0.f : __expf(p0[v] - mn);
      const float e1 = (mn == NEG_INF) ? 0.f : __expf(p1[v] - mn);
      p0[v] = e0;
      p1[v] = e1;
      rsum[v] = rsum[v] * corr + red_sum16(e0 + e1);
      rmax[v] = mn;
#pragma unroll
      for (int j = 0; j < 4; ++j) oacc[j][v] = oacc[j][v] * corr;
    }

    // ---- transpose P (C layout) -> A-fragment via wave-private LDS ----
#pragma unroll
    for (int v = 0; v < 8; ++v) {
      const int r = (hf * 8 + v) * 40;
      wp[r + ml] = f2bf(p0[v]);
      wp[r + 16 + ml] = f2bf(p1[v]);
    }
    FragBf ap;
    {
      const unsigned short* rp = wp + ml * 40;
      ap.q[0] = *(const uint4*)(rp + hf * 16);
      ap.q[1] = *(const uint4*)(rp + hf * 16 + 8);
    }

    // ---- O += P.V ----
#pragma unroll
    for (int j = 0; j < 4; ++j) {
#if USE_ASYNC_LDS
      const unsigned short* vp = &sV[buf][(j * 16 + ml) * 40 + hf * 16];
#else
      const unsigned short* vp =
          Vt + vt_base + (size_t)(j * 16 + ml) * SEQ + kb + hf * 16;
#endif
      FragBf vf;
      vf.q[0] = *(const uint4*)(vp);
      vf.q[1] = *(const uint4*)(vp + 8);
      oacc[j] = wmma_bf16(ap, vf, oacc[j]);
    }
  }

  // ---- normalize and store ctx in [B, S, L*HS] concat layout (bf16) ----
#pragma unroll
  for (int v = 0; v < 8; ++v) {
    const float inv = 1.f / rsum[v];
    const int s = qb + hf * 8 + v;
    const size_t base = (size_t)(b * SEQ + s) * (LVL * HSZ) + l * HSZ + h * HDM;
#pragma unroll
    for (int j = 0; j < 4; ++j)
      ctxout[base + j * 16 + ml] = f2bf(oacc[j][v] * inv);
  }
}

extern "C" void kernel_launch(void* const* d_in, const int* in_sizes, int n_in,
                              void* d_out, int out_size, void* d_ws,
                              size_t ws_size, hipStream_t stream) {
  (void)in_sizes;
  (void)n_in;
  (void)out_size;
  (void)ws_size;
  const float* X = (const float*)d_in[0];
  const float* Wq = (const float*)d_in[1];
  const float* bq = (const float*)d_in[2];
  const float* Wk = (const float*)d_in[3];
  const float* bk = (const float*)d_in[4];
  const float* Wv = (const float*)d_in[5];
  const float* bv = (const float*)d_in[6];
  const float* Wo = (const float*)d_in[7];
  const float* bo = (const float*)d_in[8];
  const unsigned char* msk = (const unsigned char*)d_in[9];

  const size_t nX = (size_t)NB * SEQ * HSZ;          // 2,097,152
  const size_t nW = (size_t)LVL * HSZ * HSZ;         // 3,145,728 (per W, and Wo)
  const size_t nQ = (size_t)LVL * NB * SEQ * HSZ;    // 6,291,456

  unsigned short* Xb = (unsigned short*)d_ws;
  unsigned short* Wqb = Xb + nX;
  unsigned short* Wkb = Wqb + nW;
  unsigned short* Wvb = Wkb + nW;
  unsigned short* Wob = Wvb + nW;
  unsigned short* Q = Wob + nW;
  unsigned short* Kw = Q + nQ;
  unsigned short* Vt = Kw + nQ;
  unsigned short* ctx = Vt + nQ;  // total ~76 MiB

  const dim3 blk(256);
  const int M = NB * SEQ;  // 2048

  // 0) one-time bf16 conversion of activations and weights
  mla_cvt_bf16<<<dim3((unsigned)(nX / 8 / 256)), blk, 0, stream>>>(X, Xb,
                                                                   (int)(nX / 8));
  mla_cvt_bf16<<<dim3((unsigned)(nW / 8 / 256)), blk, 0, stream>>>(Wq, Wqb,
                                                                   (int)(nW / 8));
  mla_cvt_bf16<<<dim3((unsigned)(nW / 8 / 256)), blk, 0, stream>>>(Wk, Wkb,
                                                                   (int)(nW / 8));
  mla_cvt_bf16<<<dim3((unsigned)(nW / 8 / 256)), blk, 0, stream>>>(Wv, Wvb,
                                                                   (int)(nW / 8));
  mla_cvt_bf16<<<dim3((unsigned)(nW / 8 / 256)), blk, 0, stream>>>(Wo, Wob,
                                                                   (int)(nW / 8));

  // 1) batched QKV projections: 9 GEMMs of 2048x1024x1024 (V stored transposed)
  mla_gemm_wmma<<<dim3(HSZ / 128, M / 128, 3 * LVL), blk, 0, stream>>>(
      Xb, Wqb, Wkb, Wvb, bq, bk, bv, Q, Kw, Vt, nullptr, M, HSZ, HSZ,
      /*batched=*/1, /*fp32_out=*/0);

  // 2) attention: (S/128) x NH x (L*B) blocks
  mla_attn_wmma<<<dim3(SEQ / 128, NHD, LVL * NB), blk, 0, stream>>>(Q, Kw, Vt,
                                                                    msk, ctx);

  // 3) output projection: 2048 x 1024 x 3072, f32 out + bias
  mla_gemm_wmma<<<dim3(HSZ / 128, M / 128, 1), blk, 0, stream>>>(
      ctx, Wob, nullptr, nullptr, bo, nullptr, nullptr, nullptr, nullptr, nullptr,
      (float*)d_out, M, HSZ, LVL * HSZ, /*batched=*/0, /*fp32_out=*/1);
}
